// StnlsNeighAttnAgg_87110526697932
// MI455X (gfx1250) — matrix operand order repaired and checked by reading, not compiled, not run on hardware
//
#include <hip/hip_runtime.h>

// ---------------------------------------------------------------------------
// STNLS neighbor-attention aggregation for MI455X (gfx1250, wave32, WMMA).
//   0) pre-convert v_w / proj_w f32 -> bf16 (once, tiny)
//   1) v   = x @ v_w^T          (bf16 WMMA, f32 acc)   M=65536, N=K=256
//   2) agg = sum_k attn * gather(v, flows)   (f32 gather kernel, bf16 store)
//   3) out = agg @ proj_w^T + b (bf16 WMMA, f32 acc)
// B tiles (and kernel-3 A tile) staged to LDS with async global->LDS copies
// (GLOBAL_LOAD_ASYNC_TO_LDS_B128, ASYNCcnt) when the builtin is available.
// ---------------------------------------------------------------------------

typedef __attribute__((ext_vector_type(16))) __bf16 v16bf;
typedef __attribute__((ext_vector_type(8)))  float  v8f;
typedef __attribute__((ext_vector_type(4)))  int    v4i;

#define T_  4
#define H_  128
#define W_  128
#define C_  256
#define HD_ 8
#define K_  9
#define F_  32
#define M_  (T_ * H_ * W_)   // 65536

#if defined(__has_builtin)
#if __has_builtin(__builtin_amdgcn_global_load_async_to_lds_b128) && \
    __has_builtin(__builtin_amdgcn_s_wait_asynccnt)
#define HAVE_ASYNC_LDS 1
#endif
#endif
#ifndef HAVE_ASYNC_LDS
#define HAVE_ASYNC_LDS 0
#endif

typedef __attribute__((address_space(1))) v4i* gptr_v4i;
typedef __attribute__((address_space(3))) v4i* lptr_v4i;

// Native bf16 converts (v_cvt_pk_bf16_f32 on CDNA5) instead of bit tricks.
static __device__ __forceinline__ unsigned short to_bf16u(float f) {
  union { __bf16 h; unsigned short u; } cv;
  cv.h = (__bf16)f;
  return cv.u;
}
static __device__ __forceinline__ unsigned int pack_bf16x2(float lo, float hi) {
  union { __bf16 h[2]; unsigned int u; } cv;
  cv.h[0] = (__bf16)lo; cv.h[1] = (__bf16)hi;
  return cv.u;
}

// LDS tile: 64 rows x 256 bf16 halves, XOR-swizzled at 16B-chunk granularity
// so reading one k-chunk across rows 0..15 hits 16 distinct bank groups.
static __device__ __forceinline__ int swz(int row, int khalf) {
  int chunk  = khalf >> 3;   // 8 halves = 16B
  int within = khalf & 7;
  return row * 256 + (((chunk ^ (row & 15)) << 3) | within);
}

union FragBF { uint4 q[2]; v16bf v; };

// Stage a 64x256 bf16 tile (global row stride C_) into swizzled LDS.
// 256 threads, 8 x 16B per thread. Async path bypasses VGPRs (ASYNCcnt).
static __device__ __forceinline__ void stage_tile_bf16(
    const unsigned short* __restrict__ g, unsigned short* lds, int tid) {
#pragma unroll
  for (int i = 0; i < 8; ++i) {
    int j   = i * 256 + tid;     // 0..2047 uint4 slots
    int row = j >> 5;            // 32 uint4 per row
    int q8  = j & 31;
    const unsigned short* src = g + (size_t)row * C_ + q8 * 8;
    unsigned short* dst = &lds[swz(row, q8 * 8)];
#if HAVE_ASYNC_LDS
    __builtin_amdgcn_global_load_async_to_lds_b128(
        (gptr_v4i)(v4i*)src, (lptr_v4i)(v4i*)dst, 0, 0);
#else
    *(uint4*)dst = *(const uint4*)src;
#endif
  }
}

static __device__ __forceinline__ void stage_wait() {
#if HAVE_ASYNC_LDS
  __builtin_amdgcn_s_wait_asynccnt(0);
#endif
  __syncthreads();
}

// ---------------------------------------------------------------------------
// Kernel 0: convert both 256x256 f32 weight matrices to bf16 once.
// grid 128 x 256 threads, 2 elements (one packed dword) per weight per thread.
// ---------------------------------------------------------------------------
__global__ __launch_bounds__(256)
void k_cvt_w(const float* __restrict__ wa, const float* __restrict__ wb,
             unsigned int* __restrict__ oa, unsigned int* __restrict__ ob) {
  int i = blockIdx.x * 256 + threadIdx.x;   // pair index, 32768 per matrix
  float2 a = *(const float2*)(wa + (size_t)i * 2);
  float2 b = *(const float2*)(wb + (size_t)i * 2);
  oa[i] = pack_bf16x2(a.x, a.y);
  ob[i] = pack_bf16x2(b.x, b.y);
}

// ---------------------------------------------------------------------------
// Shared WMMA compute: 64x64 tile, 8 waves, 2 accumulators per wave sharing
// one A fragment. Fragment k order: k = 16*(lane/16) + s in both A and B (bf16 16x32 A, 32x16 B).
// ---------------------------------------------------------------------------
static __device__ __forceinline__ void wmma_tile_64x64(
    const unsigned short* As, const unsigned short* Bs, int tid,
    v8f& c0, v8f& c1, int& orow_t, int& oc0_t, int& oc1_t) {
  const int wv   = tid >> 5;
  const int lane = tid & 31;
  const int r    = lane & 15;   // A row / B column within subtile
  const int hx   = lane >> 4;   // A k-half select / B k-group select
  const int sub_m  = wv & 3;
  const int sub_n0 = wv >> 2;   // 0 or 1 (second accumulator uses +2)

#pragma unroll
  for (int kk = 0; kk < 8; ++kk) {
    int k0 = kk * 32;
    FragBF a, b0, b1;
    int arow = sub_m * 16 + r;
    a.q[0] = *(const uint4*)&As[swz(arow, k0 + hx * 16)];
    a.q[1] = *(const uint4*)&As[swz(arow, k0 + hx * 16 + 8)];
    int kb = k0 + hx * 16;
    int brow0 = sub_n0 * 16 + r;
    b0.q[0] = *(const uint4*)&Bs[swz(brow0, kb)];
    b0.q[1] = *(const uint4*)&Bs[swz(brow0, kb + 8)];
    int brow1 = (sub_n0 + 2) * 16 + r;
    b1.q[0] = *(const uint4*)&Bs[swz(brow1, kb)];
    b1.q[1] = *(const uint4*)&Bs[swz(brow1, kb + 8)];
    c0 = __builtin_amdgcn_wmma_f32_16x16x32_bf16(false, a.v, false, b0.v,
                                                 (short)0, c0, false, false);
    c1 = __builtin_amdgcn_wmma_f32_16x16x32_bf16(false, a.v, false, b1.v,
                                                 (short)0, c1, false, false);
  }
  // C/D layout: lanes 0-15 -> rows 0..7, lanes 16-31 -> rows 8..15
  orow_t = sub_m * 16 + hx * 8;
  oc0_t  = sub_n0 * 16 + r;
  oc1_t  = (sub_n0 + 2) * 16 + r;
}

// ---------------------------------------------------------------------------
// Kernel 1: v[m, n] = sum_k x[m, k] * v_w[n, k]
// A: f32 -> bf16 converted while staging; B: pre-converted bf16 (async copy).
// ---------------------------------------------------------------------------
__global__ __launch_bounds__(256)
void k_vproj(const float* __restrict__ x, const unsigned short* __restrict__ vw_bf,
             float* __restrict__ v) {
  __shared__ unsigned short As[64 * 256];
  __shared__ unsigned short Bs[64 * 256];
  const int tid = threadIdx.x;
  const int m0  = blockIdx.x * 64;
  const int n0  = blockIdx.y * 64;

  stage_tile_bf16(vw_bf + (size_t)n0 * C_, Bs, tid);
#pragma unroll
  for (int i = 0; i < 16; ++i) {
    int j   = i * 256 + tid;   // 0..4095 float4 slots (64 rows x 64 f4)
    int row = j >> 6;
    int c4  = j & 63;
    float4 a = *(const float4*)(x + (size_t)(m0 + row) * C_ + c4 * 4);
    unsigned int lo = pack_bf16x2(a.x, a.y);
    unsigned int hi = pack_bf16x2(a.z, a.w);
    uint2 pa = { lo, hi };
    *(uint2*)&As[swz(row, c4 * 4)] = pa;   // 8B aligned (within = 0 or 4)
  }
  stage_wait();

  v8f c0 = {}; v8f c1 = {};
  int orow_t, oc0_t, oc1_t;
  wmma_tile_64x64(As, Bs, tid, c0, c1, orow_t, oc0_t, oc1_t);

  int orow = m0 + orow_t;
  int oc0  = n0 + oc0_t;
  int oc1  = n0 + oc1_t;
#pragma unroll
  for (int i = 0; i < 8; ++i) {
    v[(size_t)(orow + i) * C_ + oc0] = c0[i];
    v[(size_t)(orow + i) * C_ + oc1] = c1[i];
  }
}

// ---------------------------------------------------------------------------
// Kernel 2: flow-guided gather + attention-weighted sum over K neighbors.
// One block per pixel p=(t,h,w); thread = (head d = tid/32, feature f = tid&31).
// ---------------------------------------------------------------------------
__global__ __launch_bounds__(256)
void k_gather(const float* __restrict__ v, const float* __restrict__ attn,
              const int* __restrict__ flows, unsigned short* __restrict__ agg) {
  __shared__ float attn_s[HD_ * K_];        // 72
  __shared__ int   flow_s[HD_ * K_ * 3];    // 216
  const int tid = threadIdx.x;
  const int p   = blockIdx.x;
  const int t   = p / (H_ * W_);
  const int hw  = p % (H_ * W_);
  const int h   = hw / W_;
  const int w   = hw % W_;

  if (tid < HD_ * K_) {
    int d = tid / K_, k = tid % K_;
    attn_s[tid] = attn[((size_t)d * M_ + p) * K_ + k];
  }
  if (tid < HD_ * K_ * 3) {
    int d = tid / (K_ * 3), rem = tid % (K_ * 3);
    flow_s[tid] = flows[(((size_t)d * M_ + p) * K_) * 3 + rem];
  }
  __syncthreads();

  const int d = tid >> 5;
  const int f = tid & 31;
  float acc = 0.f;
#pragma unroll
  for (int k = 0; k < K_; ++k) {
    int dt = flow_s[d * 27 + k * 3 + 0];
    int dh = flow_s[d * 27 + k * 3 + 1];
    int dw = flow_s[d * 27 + k * 3 + 2];
    int tt = t + dt; tt = tt < 0 ? 0 : (tt > T_ - 1 ? T_ - 1 : tt);
    int hh = h + dh; hh = hh < 0 ? 0 : (hh > H_ - 1 ? H_ - 1 : hh);
    int ww = w + dw; ww = ww < 0 ? 0 : (ww > W_ - 1 ? W_ - 1 : ww);
    size_t vidx = (((size_t)tt * H_ + hh) * W_ + ww) * C_ + d * F_ + f;
    acc += attn_s[d * K_ + k] * v[vidx];
  }
  agg[(size_t)p * C_ + tid] = to_bf16u(acc);
}

// ---------------------------------------------------------------------------
// Kernel 3: out[m, n] = sum_k agg[m, k] * proj_w[n, k] + proj_b[n]
// Both tiles already bf16 in global -> both staged with async copies.
// ---------------------------------------------------------------------------
__global__ __launch_bounds__(256)
void k_proj(const unsigned short* __restrict__ agg,
            const unsigned short* __restrict__ pw_bf,
            const float* __restrict__ pb, float* __restrict__ out) {
  __shared__ unsigned short As[64 * 256];
  __shared__ unsigned short Bs[64 * 256];
  const int tid = threadIdx.x;
  const int m0  = blockIdx.x * 64;
  const int n0  = blockIdx.y * 64;

  stage_tile_bf16(agg + (size_t)m0 * C_, As, tid);
  stage_tile_bf16(pw_bf + (size_t)n0 * C_, Bs, tid);
  stage_wait();

  v8f c0 = {}; v8f c1 = {};
  int orow_t, oc0_t, oc1_t;
  wmma_tile_64x64(As, Bs, tid, c0, c1, orow_t, oc0_t, oc1_t);

  int orow = m0 + orow_t;
  int oc0  = n0 + oc0_t;
  int oc1  = n0 + oc1_t;
  float bias0 = pb[oc0];
  float bias1 = pb[oc1];
#pragma unroll
  for (int i = 0; i < 8; ++i) {
    out[(size_t)(orow + i) * C_ + oc0] = c0[i] + bias0;
    out[(size_t)(orow + i) * C_ + oc1] = c1[i] + bias1;
  }
}

// ---------------------------------------------------------------------------
extern "C" void kernel_launch(void* const* d_in, const int* in_sizes, int n_in,
                              void* d_out, int out_size, void* d_ws, size_t ws_size,
                              hipStream_t stream) {
  (void)in_sizes; (void)n_in; (void)out_size; (void)ws_size;
  const float* x      = (const float*)d_in[0];
  const float* attn   = (const float*)d_in[1];
  const int*   flows  = (const int*)d_in[2];
  const float* v_w    = (const float*)d_in[3];
  const float* proj_w = (const float*)d_in[4];
  const float* proj_b = (const float*)d_in[5];
  float* out = (float*)d_out;

  // Workspace layout:
  //   v      f32   [M_, C_]  at 0          (64 MB)
  //   agg    bf16  [M_, C_]  at 64 MB      (32 MB)
  //   vw_bf  bf16  [C_, C_]  at 96 MB      (128 KB)
  //   pw_bf  bf16  [C_, C_]  at 96 MB+128K (128 KB)
  char* ws = (char*)d_ws;
  float*          v     = (float*)ws;
  unsigned short* agg   = (unsigned short*)(ws + (size_t)M_ * C_ * 4);
  unsigned short* vw_bf = (unsigned short*)(ws + (size_t)M_ * C_ * 6);
  unsigned short* pw_bf = (unsigned short*)(ws + (size_t)M_ * C_ * 6 + (size_t)C_ * C_ * 2);

  k_cvt_w<<<(C_ * C_) / (256 * 2), 256, 0, stream>>>(
      v_w, proj_w, (unsigned int*)vw_bf, (unsigned int*)pw_bf);

  dim3 gemm_grid(M_ / 64, C_ / 64);   // (1024, 4)
  k_vproj <<<gemm_grid, 256, 0, stream>>>(x, vw_bf, v);
  k_gather<<<M_,        256, 0, stream>>>(v, attn, flows, agg);
  k_proj  <<<gemm_grid, 256, 0, stream>>>(agg, pw_bf, proj_b, out);
}
